// Longformer_34162169872557
// MI455X (gfx1250) — matrix operand order repaired and checked by reading, not compiled, not run on hardware
//
#include <hip/hip_runtime.h>
#include <cstdint>
#include <cstddef>

// ---------------- types / constants ----------------
typedef __bf16 bf16;
typedef __bf16 v16bf __attribute__((ext_vector_type(16)));
typedef __bf16 v8bf  __attribute__((ext_vector_type(8)));
typedef float  v8f   __attribute__((ext_vector_type(8)));

#define S_LEN  2048
#define DMODEL 768
#define NHEAD  12
#define DHEAD  64
#define FFDIM  3072
#define NLAYER 12
#define WIN    256
#define NCHUNK (S_LEN / WIN)   // 8
#define W3     (3 * WIN)       // 768

// ---------------- WMMA helpers ----------------
static __device__ __forceinline__ v16bf ld_afrag(const bf16* p) {
  // A fragment (16x32 bf16): per-lane two contiguous 8-element chunks
  v8bf lo = *(const v8bf*)p;
  v8bf hi = *(const v8bf*)(p + 8);
  v16bf a;
#pragma unroll
  for (int i = 0; i < 8; ++i) { a[i] = lo[i]; a[i + 8] = hi[i]; }
  return a;
}

static __device__ __forceinline__ v8f wmma_bf16(v16bf a, v16bf b, v8f c) {
  return __builtin_amdgcn_wmma_f32_16x16x32_bf16(false, a, false, b, (short)0, c,
                                                 false, false);
}

// ---------------- generic GEMM: C[M,N] = A[M,K](bf16) x Bt[N,K](bf16) ----------------
// block = 256 threads (8 waves, 4x2), block tile 128x64, wave tile 32x32 (2x2 WMMA)
// grid.x = N/64, grid.y = M/128  -> 192 WGs for N=768, 768 WGs for N=3072
template <bool BIAS, bool GELU_ACT, bool BF16OUT>
__global__ __launch_bounds__(256) void gemm_bf16(const bf16* __restrict__ A,
                                                 const bf16* __restrict__ Bt,
                                                 const float* __restrict__ bias,
                                                 float* __restrict__ Cf,
                                                 bf16* __restrict__ Cb,
                                                 int N, int K) {
  const int lane = threadIdx.x & 31;
  const int wid  = threadIdx.x >> 5;
  const int wm = wid & 3, wn = wid >> 2;
  const int m0 = blockIdx.y * 128 + wm * 32;
  const int n0 = blockIdx.x * 64 + wn * 32;
  const int l16  = lane & 15;
  const int hsel = lane >> 4;

  const bf16* aRow[2];
#pragma unroll
  for (int mt = 0; mt < 2; ++mt)
    aRow[mt] = A + (size_t)(m0 + mt * 16 + l16) * K + (hsel << 4);
  const bf16* bRow[2];
#pragma unroll
  for (int nt = 0; nt < 2; ++nt)
    bRow[nt] = Bt + (size_t)(n0 + nt * 16 + l16) * K + (hsel << 4);

  v8f acc[2][2];
#pragma unroll
  for (int mt = 0; mt < 2; ++mt)
#pragma unroll
    for (int nt = 0; nt < 2; ++nt) acc[mt][nt] = {};

  for (int kk = 0; kk < K; kk += 32) {
    const v16bf b0 = *(const v16bf*)(bRow[0] + kk);
    const v16bf b1 = *(const v16bf*)(bRow[1] + kk);
    const v16bf a0 = ld_afrag(aRow[0] + kk);
    acc[0][0] = wmma_bf16(a0, b0, acc[0][0]);
    acc[0][1] = wmma_bf16(a0, b1, acc[0][1]);
    const v16bf a1 = ld_afrag(aRow[1] + kk);
    acc[1][0] = wmma_bf16(a1, b0, acc[1][0]);
    acc[1][1] = wmma_bf16(a1, b1, acc[1][1]);
  }

#pragma unroll
  for (int mt = 0; mt < 2; ++mt) {
#pragma unroll
    for (int nt = 0; nt < 2; ++nt) {
      const int nn = n0 + nt * 16 + l16;
      const float bv = BIAS ? bias[nn] : 0.0f;
#pragma unroll
      for (int r = 0; r < 8; ++r) {
        const int mm = m0 + mt * 16 + r + (hsel << 3);
        float v = acc[mt][nt][r] + bv;
        if (GELU_ACT) v = 0.5f * v * (1.0f + erff(v * 0.7071067811865475f));
        if (BF16OUT) Cb[(size_t)mm * N + nn] = (bf16)v;
        else         Cf[(size_t)mm * N + nn] = v;
      }
    }
  }
}

// ---------------- weight convert + transpose: W[K,N] f32 -> Wt[N,K] bf16 ----------------
__global__ __launch_bounds__(256) void transpose_w(const float* __restrict__ W,
                                                   bf16* __restrict__ Wt,
                                                   int K, int N) {
  int idx = blockIdx.x * 256 + threadIdx.x;
  if (idx < K * N) {
    int k = idx / N;
    int n = idx - k * N;
    Wt[(size_t)n * K + k] = (bf16)W[idx];
  }
}

// ---------------- V transpose: v[S, D] bf16 -> vt[H][DHEAD][S] bf16 ----------------
__global__ __launch_bounds__(256) void v_transpose(const bf16* __restrict__ v,
                                                   bf16* __restrict__ vt) {
  int idx = blockIdx.x * 256 + threadIdx.x;
  if (idx < S_LEN * DMODEL) {
    int s = idx / DMODEL;
    int d = idx - s * DMODEL;
    vt[(size_t)d * S_LEN + s] = v[idx];  // d = h*64+dd already contiguous in vt
  }
}

// ---------------- attention pass A: scores + band mask + softmax -> P bf16 ----------------
// grid: (qg=16, c=8, h=12); block 256 (8 waves). LDS 16x768 f32 scores.
__global__ __launch_bounds__(256) void attn_scores(const bf16* __restrict__ qb,
                                                   const bf16* __restrict__ kb,
                                                   const int* __restrict__ am,
                                                   bf16* __restrict__ P) {
  const int qg = blockIdx.x, c = blockIdx.y, h = blockIdx.z;
  const int lane = threadIdx.x & 31;
  const int wid  = threadIdx.x >> 5;
  const int l16  = lane & 15;
  const int hsel = lane >> 4;
  __shared__ float sS[16][W3];

  // Q fragments for this 16-row group (K = 64 -> two 16x16x32 steps)
  const int qrow = c * WIN + qg * 16 + l16;
  const bf16* aRow = qb + (size_t)qrow * DMODEL + h * DHEAD + (hsel << 4);
  const v16bf a0 = ld_afrag(aRow);
  const v16bf a1 = ld_afrag(aRow + 32);

  // two n-tiles per iteration: interleave WMMAs to hide WMMA->VALU hazard slots
#pragma unroll
  for (int tp = 0; tp < 3; ++tp) {
    const int nA = (wid * 6 + tp * 2) * 16;
    const int nB = nA + 16;
    int gA = (c - 1) * WIN + nA + l16;
    int gB = (c - 1) * WIN + nB + l16;
    gA = (gA < 0) ? 0 : ((gA > S_LEN - 1) ? S_LEN - 1 : gA);
    gB = (gB < 0) ? 0 : ((gB > S_LEN - 1) ? S_LEN - 1 : gB);
    const bf16* bRowA = kb + (size_t)gA * DMODEL + h * DHEAD + (hsel << 4);
    const bf16* bRowB = kb + (size_t)gB * DMODEL + h * DHEAD + (hsel << 4);
    const v16bf bA0 = *(const v16bf*)bRowA;
    const v16bf bA1 = *(const v16bf*)(bRowA + 32);
    const v16bf bB0 = *(const v16bf*)bRowB;
    const v16bf bB1 = *(const v16bf*)(bRowB + 32);
    v8f accA = {}, accB = {};
    accA = wmma_bf16(a0, bA0, accA);
    accB = wmma_bf16(a0, bB0, accB);
    accA = wmma_bf16(a1, bA1, accA);
    accB = wmma_bf16(a1, bB1, accB);
    const int cA = nA + l16, cB = nB + l16;
#pragma unroll
    for (int r = 0; r < 8; ++r) {
      sS[r + (hsel << 3)][cA] = accA[r] * 0.125f;
      sS[r + (hsel << 3)][cB] = accB[r] * 0.125f;
    }
  }
  __syncthreads();

  // softmax: each wave handles 2 rows; 24 elems per lane
#pragma unroll
  for (int rr = 0; rr < 2; ++rr) {
    const int m  = wid * 2 + rr;   // 0..15
    const int qi = qg * 16 + m;    // query index within chunk, 0..255
    float vals[24];
    float mx = -3.0e38f;
#pragma unroll
    for (int t = 0; t < 24; ++t) {
      const int j = lane + t * 32;
      const int g = (c - 1) * WIN + j;
      const bool valid = (j >= qi) && (j <= qi + 2 * WIN) && (g >= 0) &&
                         (g < S_LEN) && (am[g] != 0);
      const float v = valid ? sS[m][j] : -3.0e38f;
      vals[t] = v;
      mx = fmaxf(mx, v);
    }
#pragma unroll
    for (int o = 16; o > 0; o >>= 1) mx = fmaxf(mx, __shfl_xor(mx, o, 32));
    float sum = 0.0f;
#pragma unroll
    for (int t = 0; t < 24; ++t) {
      const float e = (vals[t] > -1.0e37f) ? __expf(vals[t] - mx) : 0.0f;
      vals[t] = e;
      sum += e;
    }
#pragma unroll
    for (int o = 16; o > 0; o >>= 1) sum += __shfl_xor(sum, o, 32);
    const float inv = 1.0f / sum;
    bf16* Prow = P + ((size_t)(h * NCHUNK + c) * WIN + qi) * W3;
#pragma unroll
    for (int t = 0; t < 24; ++t) Prow[lane + t * 32] = (bf16)(vals[t] * inv);
  }
}

// ---------------- attention pass B: out = P x Vwin -> abf bf16 [S, D] ----------------
// grid: (c=8, h=12, half=2); block 256 (8 waves), wave tile 16x64
__global__ __launch_bounds__(256) void attn_out(const bf16* __restrict__ P,
                                                const bf16* __restrict__ vt,
                                                bf16* __restrict__ abf) {
  const int c = blockIdx.x, h = blockIdx.y, half = blockIdx.z;
  const int lane = threadIdx.x & 31;
  const int wid  = threadIdx.x >> 5;
  const int l16  = lane & 15;
  const int hsel = lane >> 4;
  const int m0 = half * 128 + wid * 16;

  const bf16* Pb   = P + (size_t)(h * NCHUNK + c) * WIN * W3;
  const bf16* aRow = Pb + (size_t)(m0 + l16) * W3 + (hsel << 4);
  const bf16* vtb  = vt + (size_t)h * DHEAD * S_LEN;
  const bf16* bRow[4];
#pragma unroll
  for (int nt = 0; nt < 4; ++nt)
    bRow[nt] = vtb + (size_t)(nt * 16 + l16) * S_LEN;

  v8f acc[4];
#pragma unroll
  for (int nt = 0; nt < 4; ++nt) acc[nt] = {};

  for (int kk = 0; kk < W3; kk += 32) {
    const v16bf a = ld_afrag(aRow + kk);
    int g = (c - 1) * WIN + kk + (hsel << 4);
    g = (g < 0) ? 0 : ((g > S_LEN - 16) ? S_LEN - 16 : g);  // chunk-aligned clamp; P==0 there
#pragma unroll
    for (int nt = 0; nt < 4; ++nt) {
      const v16bf b = *(const v16bf*)(bRow[nt] + g);
      acc[nt] = wmma_bf16(a, b, acc[nt]);
    }
  }

#pragma unroll
  for (int nt = 0; nt < 4; ++nt) {
    const int nn = nt * 16 + l16;
#pragma unroll
    for (int r = 0; r < 8; ++r) {
      const int mm = m0 + r + (hsel << 3);
      abf[(size_t)(c * WIN + mm) * DMODEL + h * DHEAD + nn] = (bf16)acc[nt][r];
    }
  }
}

// ---------------- LayerNorm (optional residual), writes f32 + bf16 ----------------
template <bool RES>
__global__ __launch_bounds__(256) void ln_k(const float* __restrict__ xin,
                                            const float* __restrict__ res,
                                            const float* __restrict__ g,
                                            const float* __restrict__ b,
                                            float* __restrict__ ho,
                                            bf16* __restrict__ hb) {
  const int row = blockIdx.x;
  const int t = threadIdx.x;
  __shared__ float red[8];
  float v[3];
#pragma unroll
  for (int i = 0; i < 3; ++i) {
    const int d = t + i * 256;
    float x = xin[(size_t)row * DMODEL + d];
    if (RES) x += res[(size_t)row * DMODEL + d];
    v[i] = x;
  }
  float s = v[0] + v[1] + v[2];
#pragma unroll
  for (int o = 16; o > 0; o >>= 1) s += __shfl_xor(s, o, 32);
  if ((t & 31) == 0) red[t >> 5] = s;
  __syncthreads();
  float mean = 0.0f;
#pragma unroll
  for (int i = 0; i < 8; ++i) mean += red[i];
  mean *= (1.0f / DMODEL);
  __syncthreads();
  float q = 0.0f;
#pragma unroll
  for (int i = 0; i < 3; ++i) { const float d = v[i] - mean; q += d * d; }
#pragma unroll
  for (int o = 16; o > 0; o >>= 1) q += __shfl_xor(q, o, 32);
  if ((t & 31) == 0) red[t >> 5] = q;
  __syncthreads();
  float var = 0.0f;
#pragma unroll
  for (int i = 0; i < 8; ++i) var += red[i];
  var *= (1.0f / DMODEL);
  const float rs = rsqrtf(var + 1e-5f);
#pragma unroll
  for (int i = 0; i < 3; ++i) {
    const int d = t + i * 256;
    const float y = (v[i] - mean) * rs * g[d] + b[d];
    ho[(size_t)row * DMODEL + d] = y;
    hb[(size_t)row * DMODEL + d] = (bf16)y;
  }
}

// ---------------- position-id scan (RoBERTa style, padding_idx=1) ----------------
__global__ void pos_scan(const int* __restrict__ am, int* __restrict__ pos) {
  const int lane = threadIdx.x & 31;
  const int per = S_LEN / 32;  // 64
  const int base = lane * per;
  int vals[64];
  int s = 0;
#pragma unroll
  for (int i = 0; i < 64; ++i) { vals[i] = am[base + i]; s += vals[i]; }
  int pre = s;
#pragma unroll
  for (int d = 1; d < 32; d <<= 1) {
    int o = __shfl_up(pre, d, 32);
    if (lane >= d) pre += o;
  }
  pre -= s;  // exclusive prefix
  int c = pre;
#pragma unroll
  for (int i = 0; i < 64; ++i) {
    c += vals[i];
    pos[base + i] = c * vals[i] + 1;
  }
}

// ---------------- embedding gather ----------------
__global__ __launch_bounds__(256) void embed_k(const int* __restrict__ ids,
                                               const int* __restrict__ pos,
                                               const float* __restrict__ we,
                                               const float* __restrict__ pe,
                                               const float* __restrict__ te,
                                               float* __restrict__ out) {
  const int s = blockIdx.x;
  const int id = ids[s];
  const int p = pos[s];
#pragma unroll
  for (int i = 0; i < 3; ++i) {
    const int d = threadIdx.x + i * 256;
    out[(size_t)s * DMODEL + d] =
        we[(size_t)id * DMODEL + d] + pe[(size_t)p * DMODEL + d] + te[d];
  }
}

// ---------------- final output copy: h[0, :768] ----------------
__global__ void copy_out(const float* __restrict__ h, float* __restrict__ out) {
  const int i = blockIdx.x * 256 + threadIdx.x;
  if (i < DMODEL) out[i] = h[i];
}

// ---------------- host ----------------
extern "C" void kernel_launch(void* const* d_in, const int* in_sizes, int n_in,
                              void* d_out, int out_size, void* d_ws, size_t ws_size,
                              hipStream_t stream) {
  (void)in_sizes; (void)n_in; (void)out_size; (void)ws_size;
  const int*   ids      = (const int*)d_in[0];
  const int*   am       = (const int*)d_in[1];
  const float* word_emb = (const float*)d_in[2];
  const float* pos_emb  = (const float*)d_in[3];
  const float* type_emb = (const float*)d_in[4];
  const float* emb_g    = (const float*)d_in[5];
  const float* emb_b    = (const float*)d_in[6];
  const float* Wq = (const float*)d_in[7];  const float* bq = (const float*)d_in[8];
  const float* Wk = (const float*)d_in[9];  const float* bk = (const float*)d_in[10];
  const float* Wv = (const float*)d_in[11]; const float* bv = (const float*)d_in[12];
  const float* Wo = (const float*)d_in[13]; const float* bo = (const float*)d_in[14];
  const float* g1 = (const float*)d_in[15]; const float* b1 = (const float*)d_in[16];
  const float* W1 = (const float*)d_in[17]; const float* c1 = (const float*)d_in[18];
  const float* W2 = (const float*)d_in[19]; const float* c2 = (const float*)d_in[20];
  const float* g2 = (const float*)d_in[21]; const float* b2 = (const float*)d_in[22];

  char* ws = (char*)d_ws;
  size_t off = 0;
  auto alloc = [&](size_t bytes) -> void* {
    void* p = ws + off;
    off = (off + bytes + 255) & ~(size_t)255;
    return p;
  };
  float* hbuf = (float*)alloc((size_t)S_LEN * DMODEL * 4);
  float* obuf = (float*)alloc((size_t)S_LEN * DMODEL * 4);
  bf16*  hbf  = (bf16*)alloc((size_t)S_LEN * DMODEL * 2);
  bf16*  qb   = (bf16*)alloc((size_t)S_LEN * DMODEL * 2);
  bf16*  kb   = (bf16*)alloc((size_t)S_LEN * DMODEL * 2);
  bf16*  vb   = (bf16*)alloc((size_t)S_LEN * DMODEL * 2);
  bf16*  vt   = (bf16*)alloc((size_t)DMODEL * S_LEN * 2);
  bf16*  abf  = (bf16*)alloc((size_t)S_LEN * DMODEL * 2);
  bf16*  Pbuf = (bf16*)alloc((size_t)NHEAD * NCHUNK * WIN * W3 * 2);
  bf16*  mbf  = (bf16*)alloc((size_t)S_LEN * FFDIM * 2);
  bf16*  wqt  = (bf16*)alloc((size_t)DMODEL * DMODEL * 2);
  bf16*  wkt  = (bf16*)alloc((size_t)DMODEL * DMODEL * 2);
  bf16*  wvt  = (bf16*)alloc((size_t)DMODEL * DMODEL * 2);
  bf16*  wot  = (bf16*)alloc((size_t)DMODEL * DMODEL * 2);
  bf16*  w1t  = (bf16*)alloc((size_t)DMODEL * FFDIM * 2);
  bf16*  w2t  = (bf16*)alloc((size_t)FFDIM * DMODEL * 2);
  int*   posb = (int*)alloc((size_t)S_LEN * 4);

  const dim3 blk(256);
  const int ndd = DMODEL * DMODEL;            // 589824
  const int ndf = DMODEL * FFDIM;             // 2359296
  const int gdd = (ndd + 255) / 256;
  const int gdf = (ndf + 255) / 256;

  // embeddings
  pos_scan<<<1, 32, 0, stream>>>(am, posb);
  embed_k<<<S_LEN, blk, 0, stream>>>(ids, posb, word_emb, pos_emb, type_emb, obuf);
  ln_k<false><<<S_LEN, blk, 0, stream>>>(obuf, nullptr, emb_g, emb_b, hbuf, hbf);

  for (int l = 0; l < NLAYER; ++l) {
    // weight prep (fp32 -> bf16 transposed)
    transpose_w<<<gdd, blk, 0, stream>>>(Wq + (size_t)l * ndd, wqt, DMODEL, DMODEL);
    transpose_w<<<gdd, blk, 0, stream>>>(Wk + (size_t)l * ndd, wkt, DMODEL, DMODEL);
    transpose_w<<<gdd, blk, 0, stream>>>(Wv + (size_t)l * ndd, wvt, DMODEL, DMODEL);
    transpose_w<<<gdd, blk, 0, stream>>>(Wo + (size_t)l * ndd, wot, DMODEL, DMODEL);
    transpose_w<<<gdf, blk, 0, stream>>>(W1 + (size_t)l * ndf, w1t, DMODEL, FFDIM);
    transpose_w<<<gdf, blk, 0, stream>>>(W2 + (size_t)l * ndf, w2t, FFDIM, DMODEL);

    // QKV projections (bf16 out, bias fused)
    gemm_bf16<true, false, true><<<dim3(12, 16), blk, 0, stream>>>(
        hbf, wqt, bq + (size_t)l * DMODEL, nullptr, qb, DMODEL, DMODEL);
    gemm_bf16<true, false, true><<<dim3(12, 16), blk, 0, stream>>>(
        hbf, wkt, bk + (size_t)l * DMODEL, nullptr, kb, DMODEL, DMODEL);
    gemm_bf16<true, false, true><<<dim3(12, 16), blk, 0, stream>>>(
        hbf, wvt, bv + (size_t)l * DMODEL, nullptr, vb, DMODEL, DMODEL);
    v_transpose<<<(S_LEN * DMODEL + 255) / 256, blk, 0, stream>>>(vb, vt);

    // sliding-window attention
    attn_scores<<<dim3(16, NCHUNK, NHEAD), blk, 0, stream>>>(qb, kb, am, Pbuf);
    attn_out<<<dim3(NCHUNK, NHEAD, 2), blk, 0, stream>>>(Pbuf, vt, abf);

    // output projection + residual LN
    gemm_bf16<true, false, false><<<dim3(12, 16), blk, 0, stream>>>(
        abf, wot, bo + (size_t)l * DMODEL, obuf, nullptr, DMODEL, DMODEL);
    ln_k<true><<<S_LEN, blk, 0, stream>>>(hbuf, obuf, g1 + (size_t)l * DMODEL,
                                          b1 + (size_t)l * DMODEL, hbuf, hbf);

    // FFN: GELU fused into first GEMM
    gemm_bf16<true, true, true><<<dim3(48, 16), blk, 0, stream>>>(
        hbf, w1t, c1 + (size_t)l * FFDIM, nullptr, mbf, FFDIM, DMODEL);
    gemm_bf16<true, false, false><<<dim3(12, 16), blk, 0, stream>>>(
        mbf, w2t, c2 + (size_t)l * DMODEL, obuf, nullptr, DMODEL, FFDIM);
    ln_k<true><<<S_LEN, blk, 0, stream>>>(hbuf, obuf, g2 + (size_t)l * DMODEL,
                                          b2 + (size_t)l * DMODEL, hbuf, hbf);
  }

  copy_out<<<3, blk, 0, stream>>>(hbuf, (float*)d_out);
}
